// GNNModel_32487132627634
// MI455X (gfx1250) — matrix operand order repaired and checked by reading, not compiled, not run on hardware
//
#include <hip/hip_runtime.h>

#define N_NODES 20000
#define E_EDGES 640000
#define IND 64
#define OUTD 128
#define EDGED 16

typedef __bf16 bf16_t;
typedef __attribute__((ext_vector_type(16))) __bf16 v16bf;
typedef __attribute__((ext_vector_type(8)))  __bf16 v8bf;
typedef __attribute__((ext_vector_type(8)))  float  v8f;

// ---------------------------------------------------------------------------
// CDNA5 async-load-to-LDS helpers (ASYNCcnt path)
// ---------------------------------------------------------------------------
__device__ __forceinline__ void async_load_b128(unsigned ldsOff, const void* gaddr) {
    // per-lane: LDS[ldsOff] = MEM[gaddr] (16 bytes), tracked with ASYNCcnt
    asm volatile("global_load_async_to_lds_b128 %0, %1, off"
                 :: "v"(ldsOff), "v"(gaddr)
                 : "memory");
}
__device__ __forceinline__ void wait_asynccnt0() {
    asm volatile("s_wait_asynccnt 0x0" ::: "memory");
}

// ---------------------------------------------------------------------------
// WMMA fragment helpers (wave32, V_WMMA_F32_16X16X32_BF16, D = A(16x32)*B(32x16)+C)
// ---------------------------------------------------------------------------

union FragBF { v16bf v; v8bf h[2]; };

// A fragment: 16x32 bf16 tile, row-major, row stride `ld` elements.
// lane l: row m = l&15, half h = l>>4
//   elems 0..7  -> k = kbase + h*8 + 0..7
//   elems 8..15 -> k = kbase + 16 + h*8 + 0..7
__device__ __forceinline__ v16bf load_a_frag(const bf16_t* base, int ld, int kbase) {
    int lane = threadIdx.x & 31;
    const bf16_t* r = base + (size_t)(lane & 15) * ld + kbase + (lane >> 4) * 8;
    FragBF u;
    u.h[0] = *(const v8bf*)(r);
    u.h[1] = *(const v8bf*)(r + 16);
    return u.v;
}

// B fragment from transposed, K-padded weights Wt[dout][Kpad] (row-major).
// lane l: column n = nbase + (l&15); elems 0..7 -> k = kbase + (l>>4)*8 + 0..7, elems 8..15 -> k = kbase + 16 + (l>>4)*8 + 0..7
__device__ __forceinline__ v16bf load_b_frag(const bf16_t* Wt, int ldk, int nbase, int kbase) {
    int lane = threadIdx.x & 31;
    const bf16_t* r = Wt + (size_t)(nbase + (lane & 15)) * ldk + kbase + (lane >> 4) * 8;
    FragBF u;
    u.h[0] = *(const v8bf*)(r);
    u.h[1] = *(const v8bf*)(r + 16);
    return u.v;
}

__device__ __forceinline__ v8f wmma_bf16(v16bf a, v16bf b, v8f c) {
    return __builtin_amdgcn_wmma_f32_16x16x32_bf16(false, a, false, b, (short)0, c,
                                                   false, false);
}

// D layout: lane l holds column n = l&15, rows (l>>4)*8 + r (r = vgpr index 0..7)
template<int NOUT>
__device__ __forceinline__ void store_d_relu(bf16_t* act, int nb, v8f d, const float* bias) {
    int lane = threadIdx.x & 31;
    int n = nb + (lane & 15);
    int mh = (lane >> 4) * 8;
    float bv = bias[n];
    #pragma unroll
    for (int r = 0; r < 8; ++r) {
        float v = d[r] + bv;
        v = v > 0.f ? v : 0.f;
        act[(size_t)(mh + r) * NOUT + n] = (bf16_t)v;
    }
}

// act_out[16][NOUT] = relu(act_in[16][KPAD] @ W + b), using bf16 WMMA
template<int KPAD, int NOUT>
__device__ __forceinline__ void tile_gemm_relu(const bf16_t* actIn, const bf16_t* Wt,
                                               const float* bias, bf16_t* actOut) {
    for (int nb = 0; nb < NOUT; nb += 16) {
        v8f acc = {0.f, 0.f, 0.f, 0.f, 0.f, 0.f, 0.f, 0.f};
        #pragma unroll
        for (int kb = 0; kb < KPAD; kb += 32) {
            v16bf a = load_a_frag(actIn, KPAD, kb);
            v16bf b = load_b_frag(Wt, KPAD, nb, kb);
            acc = wmma_bf16(a, b, acc);
        }
        store_d_relu<NOUT>(actOut, nb, acc, bias);
    }
}

__device__ __forceinline__ float wave_sum(float v) {
    #pragma unroll
    for (int off = 16; off > 0; off >>= 1) v += __shfl_xor(v, off, 32);
    return v;
}

// ---------------------------------------------------------------------------
// Weight prep: W[din][dout] f32 -> Wt[dout][kpad] bf16 (zero padded in K)
// ---------------------------------------------------------------------------
__global__ void wprep_kernel(const float* __restrict__ W, bf16_t* __restrict__ Wt,
                             int din, int dout, int kpad) {
    int idx = blockIdx.x * 256 + threadIdx.x;
    if (idx >= dout * kpad) return;
    int n = idx / kpad, k = idx % kpad;
    float v = (k < din) ? W[(size_t)k * dout + n] : 0.f;
    Wt[idx] = (bf16_t)v;
}

// ---------------------------------------------------------------------------
// Node preprocess: dummy substitution + LayerNorm(bn0) -> f32 + bf16 copies
// ---------------------------------------------------------------------------
__global__ void __launch_bounds__(128) node_prep_kernel(
    const float* __restrict__ xin, const float* __restrict__ dummy,
    const float* __restrict__ g, const float* __restrict__ bta,
    float* __restrict__ xLn, bf16_t* __restrict__ xBf) {
    int wave = threadIdx.x >> 5, lane = threadIdx.x & 31;
    int n = blockIdx.x * 4 + wave;
    const float* xr = xin + (size_t)n * IND;
    bool invalid = (xr[0] == -999.0f);
    float a = invalid ? dummy[lane]      : xr[lane];
    float b = invalid ? dummy[lane + 32] : xr[lane + 32];
    float mean = wave_sum(a + b) * (1.f / 64.f);
    float da = a - mean, db = b - mean;
    float var = wave_sum(da * da + db * db) * (1.f / 64.f);
    float rinv = rsqrtf(var + 1e-6f);
    float ya = da * rinv * g[lane]      + bta[lane];
    float yb = db * rinv * g[lane + 32] + bta[lane + 32];
    size_t o = (size_t)n * IND;
    xLn[o + lane] = ya;        xLn[o + lane + 32] = yb;
    xBf[o + lane] = (bf16_t)ya; xBf[o + lane + 32] = (bf16_t)yb;
}

// ---------------------------------------------------------------------------
// Degree counts (shared by all three seg-means: same dst index)
// ---------------------------------------------------------------------------
__global__ void deg_kernel(const int* __restrict__ edst, float* __restrict__ deg) {
    int e = blockIdx.x * 256 + threadIdx.x;
    if (e < E_EDGES) atomicAdd(&deg[edst[e]], 1.0f);
}

// ---------------------------------------------------------------------------
// Edge encoder: LN -> 16->128 -> relu -> 128->256 -> relu -> 256->128, *gate
// Writes enc (bf16 [E][128]) and scatter-adds into ef_sum (f32 [N][128]).
// 4 waves/block, 16 edges/wave.
// ---------------------------------------------------------------------------
__global__ void __launch_bounds__(128) edge_encoder_kernel(
    const float* __restrict__ eattr,
    const int* __restrict__ edst,
    const float* __restrict__ lnG, const float* __restrict__ lnB,
    const float* __restrict__ ec1W, const float* __restrict__ ec1b,
    const float* __restrict__ ec2W, const float* __restrict__ ec2b,
    const bf16_t* __restrict__ Wt1, const float* __restrict__ b1,
    const bf16_t* __restrict__ Wt2, const float* __restrict__ b2,
    const bf16_t* __restrict__ Wt3, const float* __restrict__ b3,
    bf16_t* __restrict__ encOut, float* __restrict__ efSum) {
    __shared__ __attribute__((aligned(16))) bf16_t sA[4][16 * 32];
    __shared__ __attribute__((aligned(16))) bf16_t sB[4][16 * 128];
    __shared__ __attribute__((aligned(16))) bf16_t sC[4][16 * 256];
    __shared__ float sEW[4][16];
    __shared__ int   sDst[4][16];

    int wave = threadIdx.x >> 5, lane = threadIdx.x & 31;
    int ebase = (blockIdx.x * 4 + wave) * 16;
    bf16_t* actA = sA[wave];
    bf16_t* actB = sB[wave];
    bf16_t* actC = sC[wave];

    __builtin_prefetch(Wt2 + (threadIdx.x << 6), 0, 1);  // global_prefetch_b8

    // stage 1: per-edge LN + edge-classifier gate (lanes 0..15, one edge each)
    if (lane < 16) {
        int e = ebase + lane;
        const float* ea = eattr + (size_t)e * EDGED;
        float v[EDGED];
        float mean = 0.f;
        #pragma unroll
        for (int i = 0; i < EDGED; ++i) { v[i] = ea[i]; mean += v[i]; }
        mean *= (1.f / EDGED);
        float var = 0.f;
        #pragma unroll
        for (int i = 0; i < EDGED; ++i) { float d = v[i] - mean; var += d * d; }
        float rinv = rsqrtf(var * (1.f / EDGED) + 1e-6f);
        #pragma unroll
        for (int i = 0; i < EDGED; ++i)
            actA[lane * 32 + i] = (bf16_t)((v[i] - mean) * rinv * lnG[i] + lnB[i]);
        #pragma unroll
        for (int i = EDGED; i < 32; ++i) actA[lane * 32 + i] = (bf16_t)0.f;
        // ew = sigmoid(ec2(relu(ec1(e_attr))))  (raw e_attr)
        float s = ec2b[0];
        #pragma unroll
        for (int j = 0; j < 16; ++j) {
            float h = ec1b[j];
            #pragma unroll
            for (int i = 0; i < 16; ++i) h += v[i] * ec1W[i * 16 + j];
            h = h > 0.f ? h : 0.f;
            s += h * ec2W[j];
        }
        sEW[wave][lane] = 1.f / (1.f + __expf(-s));
        sDst[wave][lane] = edst[e];
    }
    __syncthreads();
    tile_gemm_relu<32, 128>(actA, Wt1, b1, actB);
    __syncthreads();
    tile_gemm_relu<128, 256>(actB, Wt2, b2, actC);
    __syncthreads();
    // final: 256->128, no relu, scale by ew, store + scatter-add
    for (int nb = 0; nb < 128; nb += 16) {
        v8f acc = {0.f, 0.f, 0.f, 0.f, 0.f, 0.f, 0.f, 0.f};
        #pragma unroll
        for (int kb = 0; kb < 256; kb += 32) {
            v16bf a = load_a_frag(actC, 256, kb);
            v16bf b = load_b_frag(Wt3, 256, nb, kb);
            acc = wmma_bf16(a, b, acc);
        }
        int n = nb + (lane & 15);
        int mh = (lane >> 4) * 8;
        float bv = b3[n];
        #pragma unroll
        for (int r = 0; r < 8; ++r) {
            int m = mh + r;
            float val = (acc[r] + bv) * sEW[wave][m];
            encOut[(size_t)(ebase + m) * OUTD + n] = (bf16_t)val;
            atomicAdd(&efSum[(size_t)sDst[wave][m] * OUTD + n], val);
        }
    }
}

// ---------------------------------------------------------------------------
// EdgeMLPConv: m = [x[dst] | x[src] | enc] (16x256) -> 3-layer MLP -> scatter
// NH = 64 (conv1) or 128 (conv2). actC overlays actA (dead after layer 1).
// A-tile staging uses GLOBAL_LOAD_ASYNC_TO_LDS_B128 (ASYNCcnt) gathers:
// single __shared__ buffer so LDS byte offsets == offsets into smem.
// ---------------------------------------------------------------------------
template<int NH>
__global__ void __launch_bounds__(128) conv_kernel(
    const bf16_t* __restrict__ xNode,   // [N][64] bf16
    const bf16_t* __restrict__ encBf,   // [E][128] bf16
    const int* __restrict__ esrc, const int* __restrict__ edst,
    const bf16_t* __restrict__ Wt1, const float* __restrict__ b1,
    const bf16_t* __restrict__ Wt2, const float* __restrict__ b2,
    const bf16_t* __restrict__ Wt3, const float* __restrict__ b3,
    float* __restrict__ outSum) {       // [N][NH] f32 accumulators
    constexpr int A_BYTES = 16 * 256 * 2;        // 8 KB per-wave A tile
    constexpr int B_BYTES = 16 * NH * 2;
    constexpr int A_OFF = 0;
    constexpr int B_OFF = A_OFF + 4 * A_BYTES;
    constexpr int D_OFF = B_OFF + 4 * B_BYTES;
    __shared__ __attribute__((aligned(16))) char smem[D_OFF + 4 * 16 * (int)sizeof(int)];

    int wave = threadIdx.x >> 5, lane = threadIdx.x & 31;
    int ebase = (blockIdx.x * 4 + wave) * 16;
    bf16_t* actA = (bf16_t*)(smem + A_OFF + wave * A_BYTES);
    bf16_t* actB = (bf16_t*)(smem + B_OFF + wave * B_BYTES);
    bf16_t* actC = actA;  // overlay: actA dead after layer 1
    int* sDst = (int*)(smem + D_OFF + wave * 16 * (int)sizeof(int));

    __builtin_prefetch(Wt1 + (threadIdx.x << 6), 0, 1);

    // async gather of the A tile: lane l handles row (l&15);
    // half h==0 -> x[dst] (128B) + x[src] (128B), h==1 -> enc row (256B)
    int h = lane >> 4;
    int row = lane & 15;
    int e = ebase + row;
    int sidx = esrc[e], didx = edst[e];
    if (h == 0) sDst[row] = didx;
    const bf16_t* nodeD = xNode + (size_t)didx * IND;
    const bf16_t* nodeS = xNode + (size_t)sidx * IND;
    const bf16_t* encRow = encBf + (size_t)e * OUTD;
    unsigned rowBase = (unsigned)(A_OFF + wave * A_BYTES + row * 512);
    #pragma unroll
    for (int i = 0; i < 16; ++i) {
        const bf16_t* g0 = (i < 8) ? (nodeD + i * 8) : (nodeS + (i - 8) * 8);
        const bf16_t* g = h ? (encRow + i * 8) : g0;           // branchless: full EXEC
        unsigned loff = rowBase + (h ? 256 + i * 16 : i * 16);
        async_load_b128(loff, g);
    }
    wait_asynccnt0();
    __syncthreads();

    tile_gemm_relu<256, NH>(actA, Wt1, b1, actB);
    __syncthreads();
    tile_gemm_relu<NH, NH>(actB, Wt2, b2, actC);
    __syncthreads();
    // final layer (no activation) + scatter-add
    for (int nb = 0; nb < NH; nb += 16) {
        v8f acc = {0.f, 0.f, 0.f, 0.f, 0.f, 0.f, 0.f, 0.f};
        #pragma unroll
        for (int kb = 0; kb < NH; kb += 32) {
            v16bf a = load_a_frag(actC, NH, kb);
            v16bf b = load_b_frag(Wt3, NH, nb, kb);
            acc = wmma_bf16(a, b, acc);
        }
        int n = nb + (lane & 15);
        int mh = (lane >> 4) * 8;
        float bv = b3[n];
        #pragma unroll
        for (int r = 0; r < 8; ++r) {
            float val = acc[r] + bv;
            atomicAdd(&outSum[(size_t)sDst[mh + r] * NH + n], val);
        }
    }
}

// ---------------------------------------------------------------------------
// x1 finalize: x1 = leaky_relu(LN(x1_sum/deg), 0.01) -> bf16
// ---------------------------------------------------------------------------
__global__ void __launch_bounds__(128) x1_finalize_kernel(
    const float* __restrict__ x1s, const float* __restrict__ deg,
    const float* __restrict__ g, const float* __restrict__ bta,
    bf16_t* __restrict__ x1bf) {
    int wave = threadIdx.x >> 5, lane = threadIdx.x & 31;
    int n = blockIdx.x * 4 + wave;
    float inv = 1.f / fmaxf(deg[n], 1.f);
    size_t o = (size_t)n * IND;
    float a = x1s[o + lane] * inv;
    float b = x1s[o + lane + 32] * inv;
    float mean = wave_sum(a + b) * (1.f / 64.f);
    float da = a - mean, db = b - mean;
    float var = wave_sum(da * da + db * db) * (1.f / 64.f);
    float rinv = rsqrtf(var + 1e-6f);
    float ya = da * rinv * g[lane]      + bta[lane];
    float yb = db * rinv * g[lane + 32] + bta[lane + 32];
    ya = ya > 0.f ? ya : 0.01f * ya;
    yb = yb > 0.f ? yb : 0.01f * yb;
    x1bf[o + lane] = (bf16_t)ya;
    x1bf[o + lane + 32] = (bf16_t)yb;
}

// ---------------------------------------------------------------------------
// Node head: x2 LN+relu, gated skip, ef_mean, xfc out, np MLP -> probs
// One wave per node.
// ---------------------------------------------------------------------------
__global__ void __launch_bounds__(128) node_head_kernel(
    const float* __restrict__ xLn,   // [N][64]
    const float* __restrict__ x2s,   // [N][128] sums
    const float* __restrict__ efs,   // [N][128] sums
    const float* __restrict__ deg,
    const float* __restrict__ bn2g, const float* __restrict__ bn2b,
    const float* __restrict__ skipW, const float* __restrict__ skipB,
    const float* __restrict__ gateW, const float* __restrict__ gateB,
    const float* __restrict__ np1W, const float* __restrict__ np1B,
    const float* __restrict__ np2W, const float* __restrict__ np2B,
    const float* __restrict__ np3W, const float* __restrict__ np3B,
    float* __restrict__ out) {       // [N*256] xfc then [N] probs
    __shared__ float sXln[4][64];
    __shared__ float sSkip[4][128];
    __shared__ float sXfc[4][256];
    __shared__ float sH1[4][128];
    __shared__ float sH2[4][64];

    int wave = threadIdx.x >> 5, lane = threadIdx.x & 31;
    int n = blockIdx.x * 4 + wave;
    float inv = 1.f / fmaxf(deg[n], 1.f);

    // x2 = relu(LN(x2_sum/deg))
    float v[4], x2v[4];
    float s = 0.f;
    #pragma unroll
    for (int q = 0; q < 4; ++q) { v[q] = x2s[(size_t)n * 128 + lane + 32 * q] * inv; s += v[q]; }
    float mean = wave_sum(s) * (1.f / 128.f);
    float vs = 0.f;
    #pragma unroll
    for (int q = 0; q < 4; ++q) { float d = v[q] - mean; vs += d * d; }
    float rinv = rsqrtf(wave_sum(vs) * (1.f / 128.f) + 1e-6f);
    #pragma unroll
    for (int q = 0; q < 4; ++q) {
        int o = lane + 32 * q;
        float t = (v[q] - mean) * rinv * bn2g[o] + bn2b[o];
        x2v[q] = t > 0.f ? t : 0.f;
    }
    sXln[wave][lane]      = xLn[(size_t)n * 64 + lane];
    sXln[wave][lane + 32] = xLn[(size_t)n * 64 + lane + 32];
    __syncthreads();
    // skip = proj_skip(x_ln)
    float sk[4];
    #pragma unroll
    for (int q = 0; q < 4; ++q) {
        int o = lane + 32 * q;
        float acc = skipB[o];
        for (int j = 0; j < 64; ++j) acc += sXln[wave][j] * skipW[j * 128 + o];
        sk[q] = acc;
        sSkip[wave][o] = acc;
    }
    __syncthreads();
    // gate / xf / ef_mean -> xfc (also written to out)
    #pragma unroll
    for (int q = 0; q < 4; ++q) {
        int o = lane + 32 * q;
        float acc = gateB[o];
        for (int j = 0; j < 128; ++j) acc += sSkip[wave][j] * gateW[j * 128 + o];
        float gt = 1.f / (1.f + __expf(-acc));
        float xf = gt * sk[q] + (1.f - gt) * x2v[q];
        float ef = efs[(size_t)n * 128 + o] * inv;
        sXfc[wave][o] = xf;
        sXfc[wave][128 + o] = ef;
        out[(size_t)n * 256 + o] = xf;
        out[(size_t)n * 256 + 128 + o] = ef;
    }
    __syncthreads();
    // h = elu(np1(xfc))
    #pragma unroll
    for (int q = 0; q < 4; ++q) {
        int o = lane + 32 * q;
        float acc = np1B[o];
        for (int j = 0; j < 256; ++j) acc += sXfc[wave][j] * np1W[j * 128 + o];
        sH1[wave][o] = acc > 0.f ? acc : (__expf(acc) - 1.f);
    }
    __syncthreads();
    // h = elu(np2(h))
    #pragma unroll
    for (int q = 0; q < 2; ++q) {
        int o = lane + 32 * q;
        float acc = np2B[o];
        for (int j = 0; j < 128; ++j) acc += sH1[wave][j] * np2W[j * 64 + o];
        sH2[wave][o] = acc > 0.f ? acc : (__expf(acc) - 1.f);
    }
    __syncthreads();
    // probs = np3(h)
    float p = sH2[wave][lane] * np3W[lane] + sH2[wave][lane + 32] * np3W[lane + 32];
    p = wave_sum(p);
    if (lane == 0) out[(size_t)N_NODES * 256 + n] = p + np3B[0];
}

// ---------------------------------------------------------------------------
// Host launcher
// ---------------------------------------------------------------------------
extern "C" void kernel_launch(void* const* d_in, const int* in_sizes, int n_in,
                              void* d_out, int out_size, void* d_ws, size_t ws_size,
                              hipStream_t stream) {
    (void)in_sizes; (void)n_in; (void)out_size; (void)ws_size;
    const float* x_in   = (const float*)d_in[0];
    const int*   eidx   = (const int*)d_in[1];
    const float* e_attr = (const float*)d_in[2];
    const int* esrc = eidx;             // edge_index[0,0,:]
    const int* edst = eidx + E_EDGES;   // edge_index[0,1,:]

    // params in jax pytree (sorted-key) flatten order
    int pi = 3;
    const float* bn0_bta = (const float*)d_in[pi++];
    const float* bn0_g   = (const float*)d_in[pi++];
    const float* bn1_bta = (const float*)d_in[pi++];
    const float* bn1_g   = (const float*)d_in[pi++];
    const float* bn2_bta = (const float*)d_in[pi++];
    const float* bn2_g   = (const float*)d_in[pi++];
    const float* c11_W = (const float*)d_in[pi++]; const float* c11_b = (const float*)d_in[pi++];
    const float* c12_W = (const float*)d_in[pi++]; const float* c12_b = (const float*)d_in[pi++];
    const float* c13_W = (const float*)d_in[pi++]; const float* c13_b = (const float*)d_in[pi++];
    const float* c21_W = (const float*)d_in[pi++]; const float* c21_b = (const float*)d_in[pi++];
    const float* c22_W = (const float*)d_in[pi++]; const float* c22_b = (const float*)d_in[pi++];
    const float* c23_W = (const float*)d_in[pi++]; const float* c23_b = (const float*)d_in[pi++];
    const float* dummy = (const float*)d_in[pi++];
    const float* ec1_W = (const float*)d_in[pi++]; const float* ec1_b = (const float*)d_in[pi++];
    const float* ec2_W = (const float*)d_in[pi++]; const float* ec2_b = (const float*)d_in[pi++];
    const float* enc1_W = (const float*)d_in[pi++]; const float* enc1_b = (const float*)d_in[pi++];
    const float* enc2_W = (const float*)d_in[pi++]; const float* enc2_b = (const float*)d_in[pi++];
    const float* enc3_W = (const float*)d_in[pi++]; const float* enc3_b = (const float*)d_in[pi++];
    const float* encln_bta = (const float*)d_in[pi++];
    const float* encln_g   = (const float*)d_in[pi++];
    const float* gate_W = (const float*)d_in[pi++]; const float* gate_b = (const float*)d_in[pi++];
    const float* np1_W = (const float*)d_in[pi++]; const float* np1_b = (const float*)d_in[pi++];
    const float* np2_W = (const float*)d_in[pi++]; const float* np2_b = (const float*)d_in[pi++];
    const float* np3_W = (const float*)d_in[pi++]; const float* np3_b = (const float*)d_in[pi++];
    const float* skip_W = (const float*)d_in[pi++]; const float* skip_b = (const float*)d_in[pi++];

    // workspace carve-out
    size_t off = 0;
    auto alloc = [&](size_t bytes) -> char* {
        char* p = (char*)d_ws + off;
        off += (bytes + 255) & ~(size_t)255;
        return p;
    };
    bf16_t* x_bf   = (bf16_t*)alloc((size_t)N_NODES * IND * 2);
    float*  x_ln   = (float*)alloc((size_t)N_NODES * IND * 4);
    bf16_t* x1_bf  = (bf16_t*)alloc((size_t)N_NODES * IND * 2);
    bf16_t* enc_bf = (bf16_t*)alloc((size_t)E_EDGES * OUTD * 2);
    float*  ef_sum = (float*)alloc((size_t)N_NODES * OUTD * 4);
    float*  x1_sum = (float*)alloc((size_t)N_NODES * IND * 4);
    float*  x2_sum = (float*)alloc((size_t)N_NODES * OUTD * 4);
    float*  deg    = (float*)alloc((size_t)N_NODES * 4);
    bf16_t* Wt_enc1 = (bf16_t*)alloc(128 * 32 * 2);
    bf16_t* Wt_enc2 = (bf16_t*)alloc(256 * 128 * 2);
    bf16_t* Wt_enc3 = (bf16_t*)alloc(128 * 256 * 2);
    bf16_t* Wt_c11  = (bf16_t*)alloc(64 * 256 * 2);
    bf16_t* Wt_c12  = (bf16_t*)alloc(64 * 64 * 2);
    bf16_t* Wt_c13  = (bf16_t*)alloc(64 * 64 * 2);
    bf16_t* Wt_c21  = (bf16_t*)alloc(128 * 256 * 2);
    bf16_t* Wt_c22  = (bf16_t*)alloc(128 * 128 * 2);
    bf16_t* Wt_c23  = (bf16_t*)alloc(128 * 128 * 2);

    // zero accumulators (graph-capturable)
    hipMemsetAsync(ef_sum, 0, (size_t)N_NODES * OUTD * 4, stream);
    hipMemsetAsync(x1_sum, 0, (size_t)N_NODES * IND * 4, stream);
    hipMemsetAsync(x2_sum, 0, (size_t)N_NODES * OUTD * 4, stream);
    hipMemsetAsync(deg,    0, (size_t)N_NODES * 4, stream);

    auto wprep = [&](const float* W, bf16_t* Wt, int din, int dout, int kpad) {
        int total = dout * kpad;
        wprep_kernel<<<(total + 255) / 256, 256, 0, stream>>>(W, Wt, din, dout, kpad);
    };
    wprep(enc1_W, Wt_enc1, 16, 128, 32);
    wprep(enc2_W, Wt_enc2, 128, 256, 128);
    wprep(enc3_W, Wt_enc3, 256, 128, 256);
    wprep(c11_W, Wt_c11, 256, 64, 256);
    wprep(c12_W, Wt_c12, 64, 64, 64);
    wprep(c13_W, Wt_c13, 64, 64, 64);
    wprep(c21_W, Wt_c21, 256, 128, 256);
    wprep(c22_W, Wt_c22, 128, 128, 128);
    wprep(c23_W, Wt_c23, 128, 128, 128);

    node_prep_kernel<<<N_NODES / 4, 128, 0, stream>>>(x_in, dummy, bn0_g, bn0_bta, x_ln, x_bf);
    deg_kernel<<<E_EDGES / 256, 256, 0, stream>>>(edst, deg);

    edge_encoder_kernel<<<E_EDGES / 64, 128, 0, stream>>>(
        e_attr, edst, encln_g, encln_bta, ec1_W, ec1_b, ec2_W, ec2_b,
        Wt_enc1, enc1_b, Wt_enc2, enc2_b, Wt_enc3, enc3_b, enc_bf, ef_sum);

    conv_kernel<64><<<E_EDGES / 64, 128, 0, stream>>>(
        x_bf, enc_bf, esrc, edst, Wt_c11, c11_b, Wt_c12, c12_b, Wt_c13, c13_b, x1_sum);

    x1_finalize_kernel<<<N_NODES / 4, 128, 0, stream>>>(x1_sum, deg, bn1_g, bn1_bta, x1_bf);

    conv_kernel<128><<<E_EDGES / 64, 128, 0, stream>>>(
        x1_bf, enc_bf, esrc, edst, Wt_c21, c21_b, Wt_c22, c22_b, Wt_c23, c23_b, x2_sum);

    node_head_kernel<<<N_NODES / 4, 128, 0, stream>>>(
        x_ln, x2_sum, ef_sum, deg, bn2_g, bn2_bta,
        skip_W, skip_b, gate_W, gate_b,
        np1_W, np1_b, np2_W, np2_b, np3_W, np3_b,
        (float*)d_out);
}
